// SelectiveSSM_68556267979147
// MI455X (gfx1250) — matrix operand order repaired and checked by reading, not compiled, not run on hardware
//
#include <hip/hip_runtime.h>
#include <hip/hip_bf16.h>
#include <math.h>

// ---------------------------------------------------------------------------
// Selective SSM on MI455X (gfx1250).
// Heavy math: x@Wg^T and x@WD^T (M=32768, K=E=1024) -> bf16 WMMA w/ f32 acc.
// Operands are pre-packed into the exact v_wmma_f32_16x16x32_bf16 per-lane
// register layouts (CDNA5 ISA 7.12.2) so GEMM fragment loads are contiguous
// 32B per lane straight from L2 (entire working set < 192MB L2).
// Fused kernel: 64x32 tile per wave (16 WMMA / k-step, 32 FLOP/B per wave,
// ~85 FLOP/B per 128x128 block); k-loop unrolled x2 for cross-iteration
// load/WMMA software pipelining.
// ---------------------------------------------------------------------------

typedef __bf16 bf16_t;
typedef __attribute__((ext_vector_type(16))) __bf16 v16bf;
typedef __attribute__((ext_vector_type(8)))  float  v8f;

constexpr int B_ = 16, S_ = 2048, D_ = 1024, N_ = 16;
constexpr int M_   = B_ * S_;     // 32768 tokens
constexpr int KT_D = D_ / 32;     // 32 k-tiles for K=1024
constexpr int MT_  = M_ / 16;     // 2048 m-tiles
constexpr int CHUNKS = 64, CLEN = 32; // scan chunking (64*32 = S)

#define DEVI static __device__ __forceinline__

// A-matrix (16x32 bf16) lane/slot -> (row m, col k). ISA 7.12.2:
// lanes 0-15: K in {0..7} u {16..23}; lanes 16-31: +8.  v = slot/2.
DEVI void a_map(int lane, int i, int& m_loc, int& k_loc) {
  int v = i >> 1;
  m_loc = lane & 15;
  k_loc = ((v < 4) ? (2 * v) : (16 + 2 * (v - 4))) + (i & 1) + ((lane >> 4) << 3);
}
// B-matrix (32x16 bf16) lane/slot -> (row k, col n).
// lanes 0-15: K in {0..7} u {16..23}; lanes 16-31: +8 (same as A).
DEVI void b_map(int lane, int i, int& k_loc, int& n_loc) {
  int v = i >> 1;
  n_loc = lane & 15;
  k_loc = ((v < 4) ? (2 * v) : (16 + 2 * (v - 4))) + (i & 1) + ((lane >> 4) << 3);
}

DEVI v8f wmma_bf16(v16bf a, v16bf b, v8f c) {
  return __builtin_amdgcn_wmma_f32_16x16x32_bf16(false, a, false, b,
                                                 (short)0, c, false, false);
}

// --------------------------- packing kernels -------------------------------

// x [M, D] f32 -> A-format bf16 tiles: tile (mt,kt), 512 elems, lane-major.
__global__ void pack_x_kernel(const float* __restrict__ x, bf16_t* __restrict__ xp) {
  int t = blockIdx.x * blockDim.x + threadIdx.x;   // over M_*D_
  int tile = t >> 9, r = t & 511;
  int lane = r >> 4, i = r & 15;
  int mt = tile / KT_D, kt = tile - mt * KT_D;
  int m_loc, k_loc; a_map(lane, i, m_loc, k_loc);
  xp[t] = (bf16_t)x[(size_t)(mt * 16 + m_loc) * D_ + (kt * 32 + k_loc)];
}

// W [E, Ksrc] f32 -> B-format bf16 tiles (zero-pad k >= Ksrc). out[m,e]=sum_k A[m,k]W[e,k].
__global__ void pack_w_kernel(const float* __restrict__ w, bf16_t* __restrict__ wp,
                              int Ksrc, int KT) {
  int t = blockIdx.x * blockDim.x + threadIdx.x;   // over (E/16)*KT*512
  int tile = t >> 9, r = t & 511;
  int lane = r >> 4, i = r & 15;
  int et = tile / KT, kt = tile - et * KT;
  int k_loc, n_loc; b_map(lane, i, k_loc, n_loc);
  int k = kt * 32 + k_loc;
  float v = (k < Ksrc) ? w[(size_t)(et * 16 + n_loc) * Ksrc + k] : 0.0f;
  wp[t] = (bf16_t)v;
}

// hs [M, 16] f32 -> A-format bf16, K=16 zero-padded to 32 (single k-tile).
__global__ void pack_h_kernel(const float* __restrict__ hs, bf16_t* __restrict__ hp) {
  int t = blockIdx.x * blockDim.x + threadIdx.x;   // over MT_*512
  int mt = t >> 9, r = t & 511;
  int lane = r >> 4, i = r & 15;
  int m_loc, k_loc; a_map(lane, i, m_loc, k_loc);
  float v = (k_loc < 16) ? hs[(mt * 16 + m_loc) * 16 + k_loc] : 0.0f;
  hp[t] = (bf16_t)v;
}

// ---------------------- a/b projections (N=16 GEMMs) -----------------------
// One wave per 16-token m-tile; 2 WMMAs per k-step (shared A fragment).
__global__ __launch_bounds__(256)
void ab_proj_kernel(const bf16_t* __restrict__ xp,
                    const bf16_t* __restrict__ wap,
                    const bf16_t* __restrict__ wbp,
                    float* __restrict__ a_buf, float* __restrict__ b_buf) {
  int lane = threadIdx.x & 31, wave = threadIdx.x >> 5;
  int mt = blockIdx.x * 8 + wave;
  v8f accA = {}, accB = {};
  const bf16_t* xbase = xp + (size_t)mt * KT_D * 512 + lane * 16;
  for (int kt = 0; kt < KT_D; ++kt) {
    v16bf a  = *(const v16bf*)(xbase + kt * 512);
    v16bf bA = *(const v16bf*)(wap + kt * 512 + lane * 16);
    v16bf bB = *(const v16bf*)(wbp + kt * 512 + lane * 16);
    accA = wmma_bf16(a, bA, accA);
    accB = wmma_bf16(a, bB, accB);
  }
  int n = lane & 15, hi = (lane >> 4) * 8;
  int m0 = mt * 16;
  #pragma unroll
  for (int v = 0; v < 8; ++v) {
    int m = m0 + v + hi;                       // C layout: VGPR v -> row v (+8 hi half)
    a_buf[m * 16 + n] = tanhf(accA[v]);
    b_buf[m * 16 + n] = accB[v];
  }
}

// ------------------------- chunked parallel scan ---------------------------
// h_t = a_t*h_{t-1} + b_t. Compose per 32-step chunk, scan 64 chunk summaries
// serially per (b,n), then re-apply. Thread id: n fastest -> coalesced 64B rows.
__global__ void scan_chunk_kernel(const float* __restrict__ a_buf,
                                  const float* __restrict__ b_buf,
                                  float* __restrict__ cA, float* __restrict__ cB) {
  int t = blockIdx.x * blockDim.x + threadIdx.x;   // B_*N_*CHUNKS = 16384
  int n = t & 15, c = (t >> 4) & (CHUNKS - 1), b = t >> 10;
  float A = 1.0f, Bv = 0.0f;
  int base = (b * S_ + c * CLEN) * 16 + n;
  for (int j = 0; j < CLEN; ++j) {
    float av = a_buf[base + j * 16];
    float bv = b_buf[base + j * 16];
    Bv = av * Bv + bv;
    A *= av;
  }
  cA[(b * 16 + n) * CHUNKS + c] = A;
  cB[(b * 16 + n) * CHUNKS + c] = Bv;
}

__global__ void scan_seq_kernel(const float* __restrict__ cA,
                                const float* __restrict__ cB,
                                float* __restrict__ hstart) {
  int t = threadIdx.x;                             // 256 = B_*N_
  float h = 0.0f;
  int base = t * CHUNKS;
  for (int c = 0; c < CHUNKS; ++c) {
    hstart[base + c] = h;
    h = cA[base + c] * h + cB[base + c];
  }
}

__global__ void scan_apply_kernel(const float* __restrict__ a_buf,
                                  const float* __restrict__ b_buf,
                                  const float* __restrict__ hstart,
                                  float* __restrict__ hs) {
  int t = blockIdx.x * blockDim.x + threadIdx.x;   // 16384
  int n = t & 15, c = (t >> 4) & (CHUNKS - 1), b = t >> 10;
  float h = hstart[(b * 16 + n) * CHUNKS + c];
  int base = (b * S_ + c * CLEN) * 16 + n;
  for (int j = 0; j < CLEN; ++j) {
    h = a_buf[base + j * 16] * h + b_buf[base + j * 16];
    hs[base + j * 16] = h;
  }
}

// ---------------------- fused dual-GEMM + gate epilogue --------------------
// Block: 256 thr = 8 waves in 2(M)x4(E); block tile 128x128. Each wave:
// 64(M)x32(E) out = 4x2 WMMA tiles x {Wg,WD} accumulators; 16 WMMAs per
// 32-wide k-step against 8KB fragments (32 FLOP/B). k-loop unrolled x2 so
// the scheduler overlaps next-step loads with current WMMAs. Then hs@WC^T
// via one padded-K WMMA per 16x16 tile, sigmoid gate mix, coalesced stores.
__global__ __launch_bounds__(256)
void fused_out_kernel(const bf16_t* __restrict__ xp,
                      const bf16_t* __restrict__ wgp,
                      const bf16_t* __restrict__ wdp,
                      const bf16_t* __restrict__ hp,
                      const bf16_t* __restrict__ wcp,
                      const float* __restrict__ bD,
                      const float* __restrict__ bg,
                      float* __restrict__ y) {
  int lane = threadIdx.x & 31, wave = threadIdx.x >> 5;
  int wm = wave >> 2, we = wave & 3;
  int mt0 = blockIdx.x * 8 + wm * 4;      // 16-row m-tile index (4 per wave)
  int et0 = blockIdx.y * 8 + we * 2;      // 16-col e-tile index (2 per wave)
  const int lo = lane * 16;

  // Per-operand base pointers; inner loop addresses are base + kt*512 so the
  // unrolled pairs get immediate-offset global_load_b128.
  const bf16_t* abase[4];
  #pragma unroll
  for (int mi = 0; mi < 4; ++mi)
    abase[mi] = xp + (size_t)(mt0 + mi) * KT_D * 512 + lo;
  const bf16_t* gbase[2]; const bf16_t* dbase[2];
  #pragma unroll
  for (int ei = 0; ei < 2; ++ei) {
    gbase[ei] = wgp + (size_t)(et0 + ei) * KT_D * 512 + lo;
    dbase[ei] = wdp + (size_t)(et0 + ei) * KT_D * 512 + lo;
  }

  v8f accg[4][2] = {};
  v8f accd[4][2] = {};

  #pragma unroll 2
  for (int kt = 0; kt < KT_D; ++kt) {
    v16bf afr[4];
    #pragma unroll
    for (int mi = 0; mi < 4; ++mi)
      afr[mi] = *(const v16bf*)(abase[mi] + kt * 512);
    #pragma unroll
    for (int ei = 0; ei < 2; ++ei) {
      v16bf bgf = *(const v16bf*)(gbase[ei] + kt * 512);
      v16bf bdf = *(const v16bf*)(dbase[ei] + kt * 512);
      #pragma unroll
      for (int mi = 0; mi < 4; ++mi) {
        accg[mi][ei] = wmma_bf16(afr[mi], bgf, accg[mi][ei]);
        accd[mi][ei] = wmma_bf16(afr[mi], bdf, accd[mi][ei]);
      }
    }
  }

  // c = hs @ WC^T  (K=16 zero-padded to 32 -> single WMMA per tile)
  v8f accc[4][2] = {};
  {
    v16bf ah[4];
    #pragma unroll
    for (int mi = 0; mi < 4; ++mi)
      ah[mi] = *(const v16bf*)(hp + (size_t)(mt0 + mi) * 512 + lo);
    #pragma unroll
    for (int ei = 0; ei < 2; ++ei) {
      v16bf bc = *(const v16bf*)(wcp + (size_t)(et0 + ei) * 512 + lo);
      #pragma unroll
      for (int mi = 0; mi < 4; ++mi)
        accc[mi][ei] = wmma_bf16(ah[mi], bc, accc[mi][ei]);
    }
  }

  int nl = lane & 15, hi = (lane >> 4) * 8;
  #pragma unroll
  for (int ei = 0; ei < 2; ++ei) {
    int e = (et0 + ei) * 16 + nl;
    float bgl = bg[e], bdl = bD[e];
    #pragma unroll
    for (int mi = 0; mi < 4; ++mi) {
      int mbase = (mt0 + mi) * 16 + hi;
      #pragma unroll
      for (int v = 0; v < 8; ++v) {
        float gl = accg[mi][ei][v] + bgl;
        float g  = 1.0f / (1.0f + __expf(-gl));
        float d  = accd[mi][ei][v] + bdl;
        y[(size_t)(mbase + v) * D_ + e] = accc[mi][ei][v] * g + d * (1.0f - g);
      }
    }
  }
}

// ------------------------------- launcher ----------------------------------
extern "C" void kernel_launch(void* const* d_in, const int* in_sizes, int n_in,
                              void* d_out, int out_size, void* d_ws, size_t ws_size,
                              hipStream_t stream) {
  const float* x  = (const float*)d_in[0];
  const float* WA = (const float*)d_in[1];
  const float* WB = (const float*)d_in[2];
  const float* WC = (const float*)d_in[3];
  const float* WD = (const float*)d_in[4];
  const float* bD = (const float*)d_in[5];
  const float* Wg = (const float*)d_in[6];
  const float* bg = (const float*)d_in[7];
  float* y = (float*)d_out;

  char* ws = (char*)d_ws;
  size_t off = 0;
  auto alloc = [&](size_t bytes) { void* p = ws + off; off += (bytes + 255) & ~size_t(255); return p; };

  bf16_t* xp  = (bf16_t*)alloc((size_t)M_ * D_ * 2);        // 64 MB packed x
  bf16_t* wgp = (bf16_t*)alloc((size_t)D_ * D_ * 2);
  bf16_t* wdp = (bf16_t*)alloc((size_t)D_ * D_ * 2);
  bf16_t* wap = (bf16_t*)alloc((size_t)N_ * D_ * 2);
  bf16_t* wbp = (bf16_t*)alloc((size_t)N_ * D_ * 2);
  bf16_t* wcp = (bf16_t*)alloc((size_t)D_ * 32 * 2);        // K padded to 32
  float* a_buf  = (float*)alloc((size_t)M_ * N_ * 4);
  float* b_buf  = (float*)alloc((size_t)M_ * N_ * 4);
  float* hs     = (float*)alloc((size_t)M_ * N_ * 4);
  bf16_t* hp    = (bf16_t*)alloc((size_t)MT_ * 512 * 2);
  float* cA     = (float*)alloc((size_t)B_ * N_ * CHUNKS * 4);
  float* cB     = (float*)alloc((size_t)B_ * N_ * CHUNKS * 4);
  float* hstart = (float*)alloc((size_t)B_ * N_ * CHUNKS * 4);

  // 1) pack operands into WMMA register layouts
  pack_x_kernel<<<(M_ * D_) / 256, 256, 0, stream>>>(x, xp);
  pack_w_kernel<<<((D_ / 16) * KT_D * 512) / 256, 256, 0, stream>>>(Wg, wgp, D_, KT_D);
  pack_w_kernel<<<((D_ / 16) * KT_D * 512) / 256, 256, 0, stream>>>(WD, wdp, D_, KT_D);
  pack_w_kernel<<<((N_ / 16) * KT_D * 512) / 256, 256, 0, stream>>>(WA, wap, D_, KT_D);
  pack_w_kernel<<<((N_ / 16) * KT_D * 512) / 256, 256, 0, stream>>>(WB, wbp, D_, KT_D);
  pack_w_kernel<<<((D_ / 16) * 1 * 512) / 256, 256, 0, stream>>>(WC, wcp, N_, 1);

  // 2) a = tanh(x@WA^T), b = x@WB^T  (WMMA)
  ab_proj_kernel<<<MT_ / 8, 256, 0, stream>>>(xp, wap, wbp, a_buf, b_buf);

  // 3) chunked parallel linear-recurrence scan -> hs
  scan_chunk_kernel<<<(B_ * N_ * CHUNKS) / 256, 256, 0, stream>>>(a_buf, b_buf, cA, cB);
  scan_seq_kernel<<<1, B_ * N_, 0, stream>>>(cA, cB, hstart);
  scan_apply_kernel<<<(B_ * N_ * CHUNKS) / 256, 256, 0, stream>>>(a_buf, b_buf, hstart, hs);
  pack_h_kernel<<<(MT_ * 512) / 256, 256, 0, stream>>>(hs, hp);

  // 4) fused dual GEMM (Wg, WD) + hs@WC^T + sigmoid gate mix -> y
  fused_out_kernel<<<dim3(MT_ / 8, (D_ / 16) / 8), 256, 0, stream>>>(
      xp, wgp, wdp, hp, wcp, bD, bg, y);
}
